// Performer_82617990906591
// MI455X (gfx1250) — matrix operand order repaired and checked by reading, not compiled, not run on hardware
//
#include <hip/hip_runtime.h>
#include <hip/hip_bf16.h>
#include <math.h>

// ---------------------------------------------------------------------------
// Performer (FAVOR+) attention for MI455X / gfx1250.
// All contractions use v_wmma_f32_16x16x32_f16 (f16 in, f32 accumulate).
// B=4 S=4096 D=1024 H=16 K=64 M=128.
// Round 3: transposed LDS staging now packs K-pairs -> v_cvt_pk_f16_f32 +
// ds_store_b32 (halves DS + cvt ops in the hot loops). Double-buffered LDS
// pipeline and b128 global loads as before.
// ---------------------------------------------------------------------------

typedef _Float16 f16;
typedef __attribute__((ext_vector_type(16))) _Float16 v16h;
typedef __attribute__((ext_vector_type(8)))  _Float16 v8h;
typedef __attribute__((ext_vector_type(2)))  _Float16 v2h;
typedef __attribute__((ext_vector_type(8)))  float    v8f;
typedef __attribute__((ext_vector_type(4)))  float    v4f;

#define WMMA_F32_F16(a, b, c) \
  __builtin_amdgcn_wmma_f32_16x16x32_f16(false, (a), false, (b), (short)0, (c), false, false)

static __device__ __forceinline__ v16h frag_cat(const f16* p0, const f16* p1) {
  v8h lo = *(const v8h*)p0;
  v8h hi = *(const v8h*)p1;
  v16h r;
#pragma unroll
  for (int i = 0; i < 8; ++i) { r[i] = lo[i]; r[8 + i] = hi[i]; }
  return r;
}
// A-fragment (16x32 f16): lane half h needs K = {8h..8h+7} then {16+8h..23+8h}.
static __device__ __forceinline__ v16h fragA(const f16* row, int h) {
  return frag_cat(row + 8 * h, row + 8 * h + 16);
}
// B-fragment (32x16 f16): lane half h needs K = {8h..8h+7} then {16+8h..23+8h}.
static __device__ __forceinline__ v16h fragB(const f16* row, int h) {
  return frag_cat(row + 8 * h, row + 8 * h + 16);
}
// Pack two floats -> one b32 LDS store (v_cvt_pk_f16_f32 + ds_store_b32).
static __device__ __forceinline__ void st_pair(f16* dst, float a, float b) {
  v2h o; o[0] = (f16)a; o[1] = (f16)b;
  *(v2h*)dst = o;
}

// ---------------------------------------------------------------------------
// Generic [nrows,1024] x [1024,1024] + bias GEMM (Q/K/V projections and Wo).
// 256 thr = 8 waves (2x4), block tile 64x128, K-step 32, double-buffered LDS.
// ---------------------------------------------------------------------------
__global__ __launch_bounds__(256) void gemm1024(const float* __restrict__ A,
                                                const float* __restrict__ W,
                                                const float* __restrict__ bias,
                                                float* __restrict__ C) {
  __shared__ f16 lA[2][64 * 40];    // [row][k], stride 40 -> 16B aligned rows
  __shared__ f16 lB[2][128 * 40];   // [col][k] transposed

  const int tid = threadIdx.x;
  const int l = tid & 31, w = tid >> 5;
  const int h = l >> 4, ln = l & 15;
  const int wr = w >> 2, wc = w & 3;           // 2x4 wave grid
  const int r0 = blockIdx.x * 64;
  const int c0 = blockIdx.y * 128;

  const int arow = tid >> 2, aq = tid & 3;     // A staging: row, 8-float quad
  const int spB = tid >> 4, ngB = tid & 15;    // B staging: s-pair, 8-col group
  const float* aSrcBase = A + (size_t)(r0 + arow) * 1024 + aq * 8;
  const float* bSrcBase = W + (size_t)(spB * 2) * 1024 + c0 + ngB * 8;

  v8f acc[2][2] = {};
  v4f ra0, ra1, rb[4];   // rb[0..1]: s0 row (8 cols), rb[2..3]: s0+1 row

  auto loadRegs = [&](int kk) {
    const float* as = aSrcBase + kk;
    ra0 = *(const v4f*)as;
    ra1 = *(const v4f*)(as + 4);
    const float* bs = bSrcBase + (size_t)kk * 1024;
    rb[0] = *(const v4f*)bs;
    rb[1] = *(const v4f*)(bs + 4);
    rb[2] = *(const v4f*)(bs + 1024);
    rb[3] = *(const v4f*)(bs + 1028);
  };
  auto storeLds = [&](int p) {
    v8h o;
#pragma unroll
    for (int i = 0; i < 4; ++i) { o[i] = (f16)ra0[i]; o[4 + i] = (f16)ra1[i]; }
    *(v8h*)&lA[p][arow * 40 + aq * 8] = o;
#pragma unroll
    for (int i = 0; i < 2; ++i)
#pragma unroll
      for (int j = 0; j < 4; ++j)
        st_pair(&lB[p][(ngB * 8 + i * 4 + j) * 40 + spB * 2],
                rb[i][j], rb[i + 2][j]);
  };
  auto compute = [&](int p) {
#pragma unroll
    for (int rt = 0; rt < 2; ++rt) {
      v16h a = fragA(&lA[p][(wr * 32 + rt * 16 + ln) * 40], h);
#pragma unroll
      for (int ct = 0; ct < 2; ++ct) {
        v16h b = fragB(&lB[p][(wc * 32 + ct * 16 + ln) * 40], h);
        acc[rt][ct] = WMMA_F32_F16(a, b, acc[rt][ct]);
      }
    }
  };

  loadRegs(0);
  storeLds(0);
  __syncthreads();
  int p = 0;
  for (int kk = 32; kk < 1024; kk += 32) {
    loadRegs(kk);        // global loads in flight while we do matrix ops
    compute(p);
    storeLds(1 - p);     // waits on loadcnt, fills the other buffer
    __syncthreads();
    p ^= 1;
  }
  compute(p);

#pragma unroll
  for (int ct = 0; ct < 2; ++ct) {
    int col = c0 + wc * 32 + ct * 16 + ln;
    float bc = bias[col];
#pragma unroll
    for (int rt = 0; rt < 2; ++rt)
#pragma unroll
      for (int v = 0; v < 8; ++v) {
        int row = r0 + wr * 32 + rt * 16 + v + 8 * h;
        C[(size_t)row * 1024 + col] = acc[rt][ct][v] + bc;
      }
  }
}

// ---------------------------------------------------------------------------
// FAVOR+ feature kernel. X: [262144,64] (q or k heads), proj: [128,64].
// mode 0: query features (per-row stab).  mode 1: key pass-1 (block max only).
// mode 2: key pass-2 (global stab from gmaxp).  Per wave: 16 rows x 128 feats.
// ---------------------------------------------------------------------------
__global__ __launch_bounds__(256) void phi_kernel(const float* __restrict__ X,
                                                  const float* __restrict__ proj,
                                                  int mode,
                                                  const float* __restrict__ gmaxp,
                                                  float* __restrict__ out,
                                                  float* __restrict__ partialOut) {
  __shared__ f16 lP[128 * 72];       // proj [m][k], stride 72
  __shared__ f16 lX[8 * 16 * 72];    // per-wave x tiles
  __shared__ float diagLds[8 * 16];
  __shared__ float wmaxLds[8];

  const int tid = threadIdx.x;
  const int l = tid & 31, w = tid >> 5;
  const int h = l >> 4, ln = l & 15;
  const float xscale = 0.35355339059327373f;  // 64^-0.25
  const float mscale = 0.08838834764831845f;  // 128^-0.5

  {  // stage proj -> f16 LDS (128x64)
    int m = tid >> 1, half = tid & 1;
    const float* src = proj + m * 64 + half * 32;
#pragma unroll
    for (int i = 0; i < 4; ++i) {
      v4f x0 = *(const v4f*)(src + i * 8);
      v4f x1 = *(const v4f*)(src + i * 8 + 4);
      v8h o;
#pragma unroll
      for (int j = 0; j < 4; ++j) { o[j] = (f16)x0[j]; o[4 + j] = (f16)x1[j]; }
      *(v8h*)&lP[m * 72 + half * 32 + i * 8] = o;
    }
  }
  const int gr0 = blockIdx.x * 128 + w * 16;
  {  // stage 16 rows of x (scaled) for this wave; fused diag = 0.5*|x|^2
    int row = l >> 1, half = l & 1;
    const float* src = X + (size_t)(gr0 + row) * 64 + half * 32;
    f16* dst = &lX[w * 16 * 72 + row * 72 + half * 32];
    float ss = 0.f;
#pragma unroll
    for (int i = 0; i < 4; ++i) {
      v4f x0 = *(const v4f*)(src + i * 8);
      v4f x1 = *(const v4f*)(src + i * 8 + 4);
      v8h o;
#pragma unroll
      for (int j = 0; j < 4; ++j) {
        float a = x0[j] * xscale, b = x1[j] * xscale;
        ss += a * a + b * b;
        o[j] = (f16)a; o[4 + j] = (f16)b;
      }
      *(v8h*)(dst + i * 8) = o;
    }
    ss = 0.5f * (ss + __shfl_xor(ss, 1, 32));
    if (half == 0) diagLds[w * 16 + row] = ss;
  }
  __syncthreads();

  // wx = x . proj^T : 16x128 per wave = 8 N-tiles x 2 K-steps of WMMA
  v16h a0 = fragA(&lX[w * 16 * 72 + ln * 72 + 0], h);
  v16h a1 = fragA(&lX[w * 16 * 72 + ln * 72 + 32], h);
  v8f acc[8];
#pragma unroll
  for (int t = 0; t < 8; ++t) {
    v16h b0 = fragB(&lP[(t * 16 + ln) * 72 + 0], h);
    v16h b1 = fragB(&lP[(t * 16 + ln) * 72 + 32], h);
    v8f c = {};
    c = WMMA_F32_F16(a0, b0, c);
    c = WMMA_F32_F16(a1, b1, c);
    acc[t] = c;
  }

  // per-row maxima: reduce across the 16 lanes of each half
  float rmax[8];
#pragma unroll
  for (int v = 0; v < 8; ++v) {
    float m = acc[0][v];
#pragma unroll
    for (int t = 1; t < 8; ++t) m = fmaxf(m, acc[t][v]);
#pragma unroll
    for (int off = 1; off < 16; off <<= 1) m = fmaxf(m, __shfl_xor(m, off, 32));
    rmax[v] = m;
  }

  if (mode == 1) {  // key pass-1: block max -> partials
    float wm = rmax[0];
#pragma unroll
    for (int v = 1; v < 8; ++v) wm = fmaxf(wm, rmax[v]);
    wm = fmaxf(wm, __shfl_xor(wm, 16, 32));
    if (l == 0) wmaxLds[w] = wm;
    __syncthreads();
    if (tid == 0) {
      float bm = wmaxLds[0];
      for (int i = 1; i < 8; ++i) bm = fmaxf(bm, wmaxLds[i]);
      partialOut[blockIdx.x] = bm;
    }
    return;
  }

  const float gstab = (mode == 2) ? gmaxp[0] : 0.f;
#pragma unroll
  for (int v = 0; v < 8; ++v) {
    float stab = (mode == 2) ? gstab : rmax[v];
    float dg = diagLds[w * 16 + v + 8 * h];
    int grow = gr0 + v + 8 * h;
#pragma unroll
    for (int t = 0; t < 8; ++t)
      out[(size_t)grow * 128 + t * 16 + ln] = __expf(acc[t][v] - dg - stab) * mscale;
  }
}

__global__ __launch_bounds__(256) void reduce_max_kernel(const float* __restrict__ partial,
                                                         int n, float* __restrict__ gmax) {
  __shared__ float s[8];
  float m = -3.4e38f;
  for (int i = threadIdx.x; i < n; i += 256) m = fmaxf(m, partial[i]);
#pragma unroll
  for (int off = 1; off < 32; off <<= 1) m = fmaxf(m, __shfl_xor(m, off, 32));
  if ((threadIdx.x & 31) == 0) s[threadIdx.x >> 5] = m;
  __syncthreads();
  if (threadIdx.x == 0) {
    float r = s[0];
    for (int i = 1; i < 8; ++i) r = fmaxf(r, s[i]);
    gmax[0] = r;
  }
}

// ---------------------------------------------------------------------------
// kv[b,h,m,d] = sum_s phi_k * v ; k1[b,h,m] = sum_s phi_k.
// One WG per (b,h); 8 waves (4x2) cover 128x64; S reduced in steps of 32.
// Double-buffered, pair-packed transposed staging; k1 kept in 8 register
// partials per thread, reduced once at the end with ds_add_f32.
// ---------------------------------------------------------------------------
__global__ __launch_bounds__(256) void kv_kernel(const float* __restrict__ phiK,
                                                 const float* __restrict__ V,
                                                 float* __restrict__ KV,
                                                 float* __restrict__ K1) {
  __shared__ f16 lPhi[2][128 * 40];  // [m][s]
  __shared__ f16 lV[2][64 * 40];     // [d][s]
  __shared__ float k1lds[128];

  const int tid = threadIdx.x;
  const int l = tid & 31, w = tid >> 5;
  const int h = l >> 4, ln = l & 15;
  const int wr = w >> 1, wc = w & 1;
  const int bh = blockIdx.x, b = bh >> 4, hh = bh & 15;

  const int spP = tid >> 4, mg = tid & 15;   // phi: s-pair, 8-m group
  const int spV = tid >> 4, dgv = tid & 15;  // v:   s-pair, 4-d group
  const float* pSrcBase =
      phiK + ((size_t)((b * 4096 + spP * 2) * 16 + hh)) * 128 + mg * 8;
  const float* vSrcBase =
      V + ((size_t)((b * 4096 + spV * 2) * 16 + hh)) * 64 + dgv * 4;
  const size_t pRow = (size_t)16 * 128;        // one s step
  const size_t vRow = (size_t)16 * 64;
  const size_t pStep = 32 * pRow;              // 32 s rows per tile
  const size_t vStep = 32 * vRow;

  v8f acc[2][2] = {};
  float k1p[8];
#pragma unroll
  for (int i = 0; i < 8; ++i) k1p[i] = 0.f;
  v4f rp[4], rv[2];  // rp[0..1]: s0 (8 m), rp[2..3]: s0+1 ; rv: s0, s0+1 (4 d)

  auto loadRegs = [&](int it) {
    const float* ps = pSrcBase + (size_t)it * pStep;
    rp[0] = *(const v4f*)ps;
    rp[1] = *(const v4f*)(ps + 4);
    rp[2] = *(const v4f*)(ps + pRow);
    rp[3] = *(const v4f*)(ps + pRow + 4);
    const float* vs = vSrcBase + (size_t)it * vStep;
    rv[0] = *(const v4f*)vs;
    rv[1] = *(const v4f*)(vs + vRow);
#pragma unroll
    for (int i = 0; i < 2; ++i)
#pragma unroll
      for (int j = 0; j < 4; ++j) k1p[i * 4 + j] += rp[i][j] + rp[i + 2][j];
  };
  auto storeLds = [&](int p) {
#pragma unroll
    for (int i = 0; i < 2; ++i)
#pragma unroll
      for (int j = 0; j < 4; ++j)
        st_pair(&lPhi[p][(mg * 8 + i * 4 + j) * 40 + spP * 2],
                rp[i][j], rp[i + 2][j]);
#pragma unroll
    for (int j = 0; j < 4; ++j)
      st_pair(&lV[p][(dgv * 4 + j) * 40 + spV * 2], rv[0][j], rv[1][j]);
  };
  auto compute = [&](int p) {
#pragma unroll
    for (int rt = 0; rt < 2; ++rt) {
      v16h a = fragA(&lPhi[p][(wr * 32 + rt * 16 + ln) * 40], h);
#pragma unroll
      for (int ct = 0; ct < 2; ++ct) {
        v16h bb = fragB(&lV[p][(wc * 32 + ct * 16 + ln) * 40], h);
        acc[rt][ct] = WMMA_F32_F16(a, bb, acc[rt][ct]);
      }
    }
  };

  if (tid < 128) k1lds[tid] = 0.f;
  loadRegs(0);
  storeLds(0);
  __syncthreads();
  int p = 0;
  for (int it = 1; it < 128; ++it) {
    loadRegs(it);
    compute(p);
    storeLds(1 - p);
    __syncthreads();
    p ^= 1;
  }
  compute(p);

#pragma unroll
  for (int rt = 0; rt < 2; ++rt)
#pragma unroll
    for (int ct = 0; ct < 2; ++ct)
#pragma unroll
      for (int v = 0; v < 8; ++v) {
        int m = wr * 32 + rt * 16 + v + 8 * h;
        int d = wc * 32 + ct * 16 + ln;
        KV[((size_t)bh * 128 + m) * 64 + d] = acc[rt][ct][v];
      }
#pragma unroll
  for (int i = 0; i < 8; ++i) atomicAdd(&k1lds[mg * 8 + i], k1p[i]);
  __syncthreads();
  if (tid < 128) K1[(size_t)bh * 128 + tid] = k1lds[tid];
}

// ---------------------------------------------------------------------------
// out[b,t,h,d] = (phi_q . kv) / (phi_q . k1 + 1e-6); denominator fused into
// the phi_q staging loop. Grid (t-chunks of 128, b*h); per wave 16x64.
// ---------------------------------------------------------------------------
__global__ __launch_bounds__(256) void qkv_kernel(const float* __restrict__ phiQ,
                                                  const float* __restrict__ KV,
                                                  const float* __restrict__ K1,
                                                  float* __restrict__ outN) {
  __shared__ f16 lKv[64 * 136];       // [d][m], stride 136
  __shared__ f16 lPhi[8 * 16 * 136];  // per-wave phi_q tiles
  __shared__ float k1lds[128];
  __shared__ float denLds[8 * 16];

  const int tid = threadIdx.x;
  const int l = tid & 31, w = tid >> 5;
  const int h = l >> 4, ln = l & 15;
  const int bh = blockIdx.y, b = bh >> 4, hh = bh & 15;
  const int t0 = blockIdx.x * 128;

  {  // stage kv transposed: m-pairs packed -> ds_store_b32
    int m0 = (tid >> 2) * 2, d0 = (tid & 3) * 16;
    const float* src = KV + ((size_t)bh * 128 + m0) * 64 + d0;
#pragma unroll
    for (int i = 0; i < 4; ++i) {
      v4f x0 = *(const v4f*)(src + i * 4);
      v4f x1 = *(const v4f*)(src + 64 + i * 4);
#pragma unroll
      for (int j = 0; j < 4; ++j)
        st_pair(&lKv[(d0 + i * 4 + j) * 136 + m0], x0[j], x1[j]);
    }
    if (tid < 128) k1lds[tid] = K1[(size_t)bh * 128 + tid];
  }
  __syncthreads();

  {  // stage 16 phi_q rows per wave; fused denominator dot with k1
    int row = l >> 1, half = l & 1;
    int grow = t0 + w * 16 + row;
    const float* src = phiQ + ((size_t)((b * 4096 + grow) * 16 + hh)) * 128 + half * 64;
    f16* dst = &lPhi[w * 16 * 136 + row * 136 + half * 64];
    float p = 0.f;
#pragma unroll
    for (int i = 0; i < 8; ++i) {
      v4f x0 = *(const v4f*)(src + i * 8);
      v4f x1 = *(const v4f*)(src + i * 8 + 4);
      v8h o;
#pragma unroll
      for (int j = 0; j < 4; ++j) {
        p += x0[j] * k1lds[half * 64 + i * 8 + j] + x1[j] * k1lds[half * 64 + i * 8 + 4 + j];
        o[j] = (f16)x0[j]; o[4 + j] = (f16)x1[j];
      }
      *(v8h*)(dst + i * 8) = o;
    }
    p += __shfl_xor(p, 1, 32);
    if (half == 0) denLds[w * 16 + row] = p;
  }

  v16h aF[4];
#pragma unroll
  for (int ks = 0; ks < 4; ++ks)
    aF[ks] = fragA(&lPhi[w * 16 * 136 + ln * 136 + 32 * ks], h);
#pragma unroll
  for (int ct = 0; ct < 4; ++ct) {
    v8f c = {};
#pragma unroll
    for (int ks = 0; ks < 4; ++ks) {
      v16h bb = fragB(&lKv[(ct * 16 + ln) * 136 + 32 * ks], h);
      c = WMMA_F32_F16(aF[ks], bb, c);
    }
#pragma unroll
    for (int v = 0; v < 8; ++v) {
      int rloc = v + 8 * h;
      float den = denLds[w * 16 + rloc];
      int grow = t0 + w * 16 + rloc;
      outN[((size_t)((b * 4096 + grow) * 16 + hh)) * 64 + ct * 16 + ln] =
          c[v] * (1.0f / (den + 1e-6f));
    }
  }
}

// ---------------------------------------------------------------------------
extern "C" void kernel_launch(void* const* d_in, const int* in_sizes, int n_in,
                              void* d_out, int out_size, void* d_ws, size_t ws_size,
                              hipStream_t stream) {
  (void)in_sizes; (void)n_in; (void)out_size; (void)ws_size;
  const float* query = (const float*)d_in[0];
  const float* value = (const float*)d_in[1];
  const float* key   = (const float*)d_in[2];
  const float* Wq = (const float*)d_in[3];
  const float* bq = (const float*)d_in[4];
  const float* Wk = (const float*)d_in[5];
  const float* bk = (const float*)d_in[6];
  const float* Wv = (const float*)d_in[7];
  const float* bv = (const float*)d_in[8];
  const float* Wo = (const float*)d_in[9];
  const float* bo = (const float*)d_in[10];
  const float* proj = (const float*)d_in[11];
  float* out = (float*)d_out;

  float* ws = (float*)d_ws;
  const size_t NRD  = (size_t)16384 * 1024;   // q/k/v head buffers
  const size_t NPHI = (size_t)262144 * 128;   // phi buffers
  float* qB   = ws; ws += NRD;
  float* kB   = ws; ws += NRD;
  float* vB   = ws; ws += NRD;
  float* phiQ = ws; ws += NPHI;
  float* phiK = ws; ws += NPHI;
  float* KV   = ws; ws += (size_t)64 * 128 * 64;
  float* K1   = ws; ws += (size_t)64 * 128;
  float* part = ws; ws += 2048;
  float* gmax = ws; ws += 1;
  float* outN = kB;  // k heads dead after phi_k pass-2 -> reuse

  dim3 blk(256);
  dim3 gG(256, 8);
  gemm1024<<<gG, blk, 0, stream>>>(query, Wq, bq, qB);
  gemm1024<<<gG, blk, 0, stream>>>(key,   Wk, bk, kB);
  gemm1024<<<gG, blk, 0, stream>>>(value, Wv, bv, vB);
  phi_kernel<<<dim3(2048), blk, 0, stream>>>(qB, proj, 0, (const float*)nullptr, phiQ, (float*)nullptr);
  phi_kernel<<<dim3(2048), blk, 0, stream>>>(kB, proj, 1, (const float*)nullptr, (float*)nullptr, part);
  reduce_max_kernel<<<dim3(1), blk, 0, stream>>>(part, 2048, gmax);
  phi_kernel<<<dim3(2048), blk, 0, stream>>>(kB, proj, 2, gmax, phiK, (float*)nullptr);
  kv_kernel<<<dim3(64), blk, 0, stream>>>(phiK, vB, KV, K1);
  qkv_kernel<<<dim3(32, 64), blk, 0, stream>>>(phiQ, KV, K1, outN);
  gemm1024<<<gG, blk, 0, stream>>>(outN, Wo, bo, out);
}
